// Channel_Transposed_Attention_12214886990453
// MI455X (gfx1250) — matrix operand order repaired and checked by reading, not compiled, not run on hardware
//
#include <hip/hip_runtime.h>

typedef __attribute__((ext_vector_type(16))) _Float16 v16h;
typedef __attribute__((ext_vector_type(8)))  _Float16 h8;
typedef __attribute__((ext_vector_type(8)))  float    v8f;

union AFrag { v16h v; h8 h[2]; };

#define DB 8
#define DC 384
#define DHH 64
#define DWW 64
#define DN 4096
#define DBN 32768
#define DC3 1152
#define DC6 64
#define DC2 192
#define DC4 96
#define DHEADS 8
#define DHD 48

// ---------------- f32 -> f16 convert ----------------
__global__ void k_cvt_f32_f16(const float* __restrict__ s, _Float16* __restrict__ d, long n) {
    long i = (long)blockIdx.x * blockDim.x + threadIdx.x;
    if (i < n) d[i] = (_Float16)s[i];
}

// ---------------- tiled transposes: dst[c*lddst + r] = src[r*ldsrc + c] ----------------
__global__ void k_transpose_f32_f16(const float* __restrict__ src, _Float16* __restrict__ dst,
                                    int R, int Ccnt, int ldsrc, int lddst, long sSrc, long sDst) {
    src += (long)blockIdx.z * sSrc;
    dst += (long)blockIdx.z * sDst;
    __shared__ float tile[32][33];
    int r0 = blockIdx.y * 32, c0 = blockIdx.x * 32;
    int tx = threadIdx.x, ty = threadIdx.y;
#pragma unroll
    for (int i = 0; i < 4; ++i) {
        int r = r0 + ty + i * 8, c = c0 + tx;
        tile[ty + i * 8][tx] = (r < R && c < Ccnt) ? src[(long)r * ldsrc + c] : 0.f;
    }
    __syncthreads();
#pragma unroll
    for (int i = 0; i < 4; ++i) {
        int c = c0 + ty + i * 8, r = r0 + tx;
        if (c < Ccnt && r < R) dst[(long)c * lddst + r] = (_Float16)tile[tx][ty + i * 8];
    }
}

__global__ void k_transpose_f16_f16(const _Float16* __restrict__ src, _Float16* __restrict__ dst,
                                    int R, int Ccnt, int ldsrc, int lddst, long sSrc, long sDst) {
    src += (long)blockIdx.z * sSrc;
    dst += (long)blockIdx.z * sDst;
    __shared__ _Float16 tile[32][33];
    int r0 = blockIdx.y * 32, c0 = blockIdx.x * 32;
    int tx = threadIdx.x, ty = threadIdx.y;
#pragma unroll
    for (int i = 0; i < 4; ++i) {
        int r = r0 + ty + i * 8, c = c0 + tx;
        tile[ty + i * 8][tx] = (r < R && c < Ccnt) ? src[(long)r * ldsrc + c] : (_Float16)0.f;
    }
    __syncthreads();
#pragma unroll
    for (int i = 0; i < 4; ++i) {
        int c = c0 + ty + i * 8, r = r0 + tx;
        if (c < Ccnt && r < R) dst[(long)c * lddst + r] = tile[tx][ty + i * 8];
    }
}

// ---------------- generic batched WMMA GEMM ----------------
// OUT[m][n] = sum_k A[m][k] * W[n][k] (+bias[n]).  A row-major MxK, W row-major NxK.
// blockIdx.z -> (zo,zi) batch; outMode: 0 = f16 row-major, 1 = f32 row-major,
// 2 = f32 transposed to (b, n_col, hw) with row m = b*HWsz + hw (final projection).
__global__ __launch_bounds__(256) void k_gemm_wmma(
    const _Float16* __restrict__ A, const _Float16* __restrict__ W,
    void* __restrict__ OUT, const float* __restrict__ bias,
    int M, int Nn, int K, int lda, int ldw, int ldo,
    int innerCnt, long sAo, long sAi, long sWo, long sWi, long sOo, long sOi,
    int outMode, int HWsz)
{
    int z = blockIdx.z;
    int zo = z / innerCnt, zi = z - zo * innerCnt;
    A += zo * sAo + zi * sAi;
    W += zo * sWo + zi * sWi;
    long outOff = zo * sOo + zi * sOi;

    __shared__ _Float16 As[128 * 40];
    __shared__ _Float16 Ws[64 * 40];

    int tid = threadIdx.x;
    int mBase = blockIdx.y * 128;
    int nBase = blockIdx.x * 64;

    int wave = tid >> 5, lane = tid & 31;
    int wm = (wave & 3) * 32, wn = (wave >> 2) * 32;
    int lhalf = lane >> 4;
    int l16 = lane & 15;

    v8f acc[2][2] = {};

    int arow = tid >> 1;
    int acol = (tid & 1) * 16;
    int wrow = tid >> 2;
    int wcol = (tid & 3) * 8;

    int ktiles = (K + 31) >> 5;
    for (int kt = 0; kt < ktiles; ++kt) {
        int k0 = kt << 5;
#pragma unroll
        for (int r = 0; r < 2; ++r) {
            int col = acol + r * 8;
            h8 v = {};
            int gm = mBase + arow, gk = k0 + col;
            if (gm < M && gk < K) v = *(const h8*)(A + (long)gm * lda + gk);
            *(h8*)(As + arow * 40 + col) = v;
        }
        {
            h8 v = {};
            int gn = nBase + wrow, gk = k0 + wcol;
            if (gn < Nn && gk < K) v = *(const h8*)(W + (long)gn * ldw + gk);
            *(h8*)(Ws + wrow * 40 + wcol) = v;
        }
        if (kt + 1 < ktiles && mBase + arow < M)
            __builtin_prefetch(A + (long)(mBase + arow) * lda + k0 + 32, 0, 1);
        __syncthreads();

        AFrag af[2], bf[2];
#pragma unroll
        for (int mi = 0; mi < 2; ++mi) {
            int row = wm + mi * 16 + l16;
            int ko = lhalf * 8;                       // lanes16-31: K 8..15 / 24..31
            af[mi].h[0] = *(const h8*)(As + row * 40 + ko);
            af[mi].h[1] = *(const h8*)(As + row * 40 + ko + 16);
        }
#pragma unroll
        for (int ni = 0; ni < 2; ++ni) {
            int colr = wn + ni * 16 + l16;
            int ko = lhalf * 8;                       // lanes16-31: K 8..15 / 24..31
            bf[ni].h[0] = *(const h8*)(Ws + colr * 40 + ko);
            bf[ni].h[1] = *(const h8*)(Ws + colr * 40 + ko + 16);
        }
#pragma unroll
        for (int mi = 0; mi < 2; ++mi)
#pragma unroll
            for (int ni = 0; ni < 2; ++ni)
                acc[mi][ni] = __builtin_amdgcn_wmma_f32_16x16x32_f16(
                    false, af[mi].v, false, bf[ni].v, (short)0, acc[mi][ni], false, false);
        __syncthreads();
    }

#pragma unroll
    for (int mi = 0; mi < 2; ++mi)
#pragma unroll
        for (int ni = 0; ni < 2; ++ni)
#pragma unroll
            for (int r = 0; r < 8; ++r) {
                int gm = mBase + wm + mi * 16 + r + 8 * lhalf;
                int gn = nBase + wn + ni * 16 + l16;
                if (gm < M && gn < Nn) {
                    float v = acc[mi][ni][r];
                    if (bias) v += bias[gn];
                    if (outMode == 0)
                        ((_Float16*)OUT)[outOff + (long)gm * ldo + gn] = (_Float16)v;
                    else if (outMode == 1)
                        ((float*)OUT)[outOff + (long)gm * ldo + gn] = v;
                    else {
                        int bb = gm / HWsz, nn = gm - bb * HWsz;
                        ((float*)OUT)[((long)bb * Nn + gn) * HWsz + nn] = v;
                    }
                }
            }
}

// ---------------- 1/max(||col||,eps) over n for q|k channels ----------------
__global__ void k_qk_norms(const _Float16* __restrict__ qkv, float* __restrict__ rn) {
    int b = blockIdx.x, c = threadIdx.x;          // blockDim = 768
    const _Float16* p = qkv + (long)b * DN * DC3 + c;
    float s = 0.f;
    for (int n = 0; n < DN; ++n) { float v = (float)p[(long)n * DC3]; s += v * v; }
    rn[b * 768 + c] = 1.f / fmaxf(sqrtf(s), 1e-12f);
}

// ---------------- softmax over d of scaled gram ----------------
__global__ void k_softmax48(const float* __restrict__ gram, const float* __restrict__ rn,
                            const float* __restrict__ temp, _Float16* __restrict__ attn) {
    int z = blockIdx.x;                           // b*8+h
    int b = z >> 3, h = z & 7;
    int c = threadIdx.x;
    if (c >= DHD) return;
    const float* g = gram + (long)z * DHD * DHD + (long)c * DHD;
    float rq = rn[b * 768 + h * DHD + c];
    float t = temp[h];
    const float* rkp = rn + b * 768 + DC + h * DHD;
    float row[DHD];
    float mx = -3.0e38f;
    for (int d = 0; d < DHD; ++d) {
        float v = g[d] * rq * rkp[d] * t;
        row[d] = v;
        mx = fmaxf(mx, v);
    }
    float s = 0.f;
    for (int d = 0; d < DHD; ++d) { float e = expf(row[d] - mx); row[d] = e; s += e; }
    float inv = 1.f / s;
    _Float16* o = attn + (long)z * DHD * DHD + (long)c * DHD;
    for (int d = 0; d < DHD; ++d) o[d] = (_Float16)(row[d] * inv);
}

// ---------------- depthwise conv on (B,H,W,C) ----------------
__global__ void k_dwconv(const _Float16* __restrict__ in, const float* __restrict__ w,
                         const float* __restrict__ bias, _Float16* __restrict__ out,
                         int Cch, int ksz, int pad, int dil, long total) {
    long id = (long)blockIdx.x * blockDim.x + threadIdx.x;
    if (id >= total) return;
    int c = (int)(id % Cch);
    long t1 = id / Cch;
    int x = (int)(t1 % DWW);
    long t2 = t1 / DWW;
    int y = (int)(t2 % DHH);
    int b = (int)(t2 / DHH);
    float acc = bias[c];
    for (int ky = 0; ky < ksz; ++ky) {
        int yy = y - pad + ky * dil;
        if (yy < 0 || yy >= DHH) continue;
        for (int kx = 0; kx < ksz; ++kx) {
            int xx = x - pad + kx * dil;
            if (xx < 0 || xx >= DWW) continue;
            acc += (float)in[(((long)b * DHH + yy) * DWW + xx) * Cch + c] * w[(c * ksz + ky) * ksz + kx];
        }
    }
    out[id] = (_Float16)acc;
}

// ---------------- mean over n, per (b,c) ----------------
__global__ void k_mean_n(const _Float16* __restrict__ in, float* __restrict__ out, int Cch) {
    int b = blockIdx.x, c = threadIdx.x;
    const _Float16* p = in + (long)b * DN * Cch + c;
    float s = 0.f;
    for (int n = 0; n < DN; ++n) s += (float)p[(long)n * Cch];
    out[b * Cch + c] = s * (1.f / (float)DN);
}

// ---------------- tiny 1x1 conv on (B,C6) means ----------------
__global__ void k_ci1(const float* __restrict__ tmean, const float* __restrict__ w,
                      const float* __restrict__ bias, float* __restrict__ out) {
    int id = threadIdx.x;                         // 512 = B*C6
    int b = id >> 6, oc = id & 63;
    float s = bias[oc];
    for (int ic = 0; ic < DC6; ++ic) s += tmean[b * DC6 + ic] * w[oc * DC6 + ic];
    out[id] = s;
}

// ---------------- t * ci1 * ci2 ----------------
__global__ void k_gprod(const _Float16* __restrict__ t, const float* __restrict__ ci1,
                        const _Float16* __restrict__ ci2, _Float16* __restrict__ out, long total) {
    long id = (long)blockIdx.x * blockDim.x + threadIdx.x;
    if (id >= total) return;
    int c = (int)(id % DC6);
    int b = (int)(id / ((long)DN * DC6));
    out[id] = (_Float16)((float)t[id] * ci1[b * DC6 + c] * (float)ci2[id]);
}

__global__ void k_add_inplace(_Float16* __restrict__ a, const _Float16* __restrict__ b, long n) {
    long id = (long)blockIdx.x * blockDim.x + threadIdx.x;
    if (id < n) a[id] = (_Float16)((float)a[id] + (float)b[id]);
}

// ---------------- gelu(x1)*x2 gate (channel split of d) ----------------
__global__ void k_gated(const _Float16* __restrict__ d, _Float16* __restrict__ gt, long total) {
    long id = (long)blockIdx.x * blockDim.x + threadIdx.x;
    if (id >= total) return;
    int j = (int)(id % DC4);
    long bn = id / DC4;
    float x1 = (float)d[bn * DC2 + j];
    float x2 = (float)d[bn * DC2 + DC4 + j];
    float g = 0.5f * x1 * (1.f + erff(x1 * 0.70710678118654752440f));
    gt[id] = (_Float16)(g * x2);
}

// ---------------- att*sig(smap) + convx*sig(cm) ----------------
__global__ void k_combine(_Float16* __restrict__ att, const _Float16* __restrict__ smap,
                          const _Float16* __restrict__ convx, const float* __restrict__ cm, long total) {
    long id = (long)blockIdx.x * blockDim.x + threadIdx.x;
    if (id >= total) return;
    int c = (int)(id % DC);
    int b = (int)(id / ((long)DN * DC));
    float s1 = 1.f / (1.f + expf(-(float)smap[id]));
    float s2 = 1.f / (1.f + expf(-cm[b * DC + c]));
    att[id] = (_Float16)((float)att[id] * s1 + (float)convx[id] * s2);
}

extern "C" void kernel_launch(void* const* d_in, const int* in_sizes, int n_in,
                              void* d_out, int out_size, void* d_ws, size_t ws_size,
                              hipStream_t stream) {
    (void)in_sizes; (void)n_in; (void)out_size; (void)ws_size;
    const float* x           = (const float*)d_in[0];
    const float* temperature = (const float*)d_in[1];
    const float* qkv_w  = (const float*)d_in[2];
    const float* proj_w = (const float*)d_in[3];
    const float* proj_b = (const float*)d_in[4];
    const float* dw1_w  = (const float*)d_in[5];
    const float* dw1_b  = (const float*)d_in[6];
    const float* dw2_w  = (const float*)d_in[7];
    const float* dw2_b  = (const float*)d_in[8];
    const float* cp_in_w  = (const float*)d_in[9];
    const float* cp_in_b  = (const float*)d_in[10];
    const float* ci1_w  = (const float*)d_in[11];
    const float* ci1_b  = (const float*)d_in[12];
    const float* ci2a_w = (const float*)d_in[13];
    const float* ci2a_b = (const float*)d_in[14];
    const float* ci2b_w = (const float*)d_in[15];
    const float* ci2b_b = (const float*)d_in[16];
    const float* ci2c_w = (const float*)d_in[17];
    const float* ci2c_b = (const float*)d_in[18];
    const float* cp_out_w = (const float*)d_in[19];
    const float* cp_out_b = (const float*)d_in[20];
    const float* sp_in_w  = (const float*)d_in[21];
    const float* sp_in_b  = (const float*)d_in[22];
    const float* sp_dw_w  = (const float*)d_in[23];
    const float* sp_dw_b  = (const float*)d_in[24];
    const float* sp_out_w = (const float*)d_in[25];
    const float* sp_out_b = (const float*)d_in[26];

    char* ws = (char*)d_ws;
    size_t o = 0;
    auto aln = [](size_t v) { return (v + 255) & ~(size_t)255; };
    const size_t OFF_R0    = o; o += aln((size_t)DB * 768 * DN * 2);   // XF16, then QT
    const size_t OFF_R1    = o; o += aln((size_t)DBN * DC3 * 2);       // QKV, then SP/SPD/GT/SMAP
    const size_t OFF_RN    = o; o += aln((size_t)DB * 768 * 4);
    const size_t OFF_GRAM  = o; o += aln((size_t)64 * DHD * DHD * 4);
    const size_t OFF_ATTN  = o; o += aln((size_t)64 * DHD * DHD * 2);
    const size_t OFF_ATT   = o; o += aln((size_t)DBN * DC * 2);
    const size_t OFF_R6    = o; o += aln((size_t)DBN * DC * 2);        // CONV1, then T/CI2A/CI2B/CI2C/GPROD
    const size_t OFF_CONVX = o; o += aln((size_t)DBN * DC * 2);
    const size_t OFF_CMAP  = o; o += aln((size_t)DBN * DC * 2);
    const size_t OFF_TMEAN = o; o += aln((size_t)DB * DC6 * 4);
    const size_t OFF_CI1   = o; o += aln((size_t)DB * DC6 * 4);
    const size_t OFF_CM    = o; o += aln((size_t)DB * DC * 4);
    const size_t OFF_WQKV  = o; o += aln((size_t)DC3 * DC * 2);
    const size_t OFF_WDW1  = o; o += aln((size_t)DC * DC * 2);
    const size_t OFF_WCPIN = o; o += aln((size_t)DC6 * DC * 2);
    const size_t OFF_WCI2C = o; o += aln((size_t)DC6 * DC6 * 2);
    const size_t OFF_WCPOUT= o; o += aln((size_t)DC * DC6 * 2);
    const size_t OFF_WSPIN = o; o += aln((size_t)DC2 * DC * 2);
    const size_t OFF_WSPOUT= o; o += aln((size_t)DC * DC4 * 2);
    const size_t OFF_WPROJ = o; o += aln((size_t)DC * DC * 2);

    _Float16* XF16 = (_Float16*)(ws + OFF_R0);
    _Float16* QT   = (_Float16*)(ws + OFF_R0);                 // reuse after XF16 dead
    _Float16* QKV  = (_Float16*)(ws + OFF_R1);
    _Float16* SP   = (_Float16*)(ws + OFF_R1);                 // reuse after QKV dead
    _Float16* SPD  = (_Float16*)(ws + OFF_R1 + (size_t)DBN * DC2 * 2);
    _Float16* GT   = (_Float16*)(ws + OFF_R1 + (size_t)2 * DBN * DC2 * 2);
    _Float16* SMAP = (_Float16*)(ws + OFF_R1 + (size_t)2 * DBN * DC2 * 2 + (size_t)DBN * DC4 * 2);
    float*    RN   = (float*)(ws + OFF_RN);
    float*    GRAM = (float*)(ws + OFF_GRAM);
    _Float16* ATTN = (_Float16*)(ws + OFF_ATTN);
    _Float16* ATT  = (_Float16*)(ws + OFF_ATT);
    _Float16* CONV1= (_Float16*)(ws + OFF_R6);
    _Float16* T16  = (_Float16*)(ws + OFF_R6);                 // reuse after CONV1 dead
    _Float16* CI2A = (_Float16*)(ws + OFF_R6 + (size_t)DBN * DC6 * 2);
    _Float16* CI2B = (_Float16*)(ws + OFF_R6 + (size_t)2 * DBN * DC6 * 2);
    _Float16* CI2C = (_Float16*)(ws + OFF_R6 + (size_t)3 * DBN * DC6 * 2);
    _Float16* GPROD= (_Float16*)(ws + OFF_R6 + (size_t)4 * DBN * DC6 * 2);
    _Float16* CONVX= (_Float16*)(ws + OFF_CONVX);
    _Float16* CMAP = (_Float16*)(ws + OFF_CMAP);
    float*    TMEAN= (float*)(ws + OFF_TMEAN);
    float*    CI1v = (float*)(ws + OFF_CI1);
    float*    CM   = (float*)(ws + OFF_CM);
    _Float16* WQKV  = (_Float16*)(ws + OFF_WQKV);
    _Float16* WDW1  = (_Float16*)(ws + OFF_WDW1);
    _Float16* WCPIN = (_Float16*)(ws + OFF_WCPIN);
    _Float16* WCI2C = (_Float16*)(ws + OFF_WCI2C);
    _Float16* WCPOUT= (_Float16*)(ws + OFF_WCPOUT);
    _Float16* WSPIN = (_Float16*)(ws + OFF_WSPIN);
    _Float16* WSPOUT= (_Float16*)(ws + OFF_WSPOUT);
    _Float16* WPROJ = (_Float16*)(ws + OFF_WPROJ);

    auto cvt = [&](const float* s, _Float16* d, long n) {
        k_cvt_f32_f16<<<(unsigned)((n + 255) / 256), 256, 0, stream>>>(s, d, n);
    };
    cvt(qkv_w, WQKV, (long)DC3 * DC);
    cvt(dw1_w, WDW1, (long)DC * DC);
    cvt(cp_in_w, WCPIN, (long)DC6 * DC);
    cvt(ci2c_w, WCI2C, (long)DC6 * DC6);
    cvt(cp_out_w, WCPOUT, (long)DC * DC6);
    cvt(sp_in_w, WSPIN, (long)DC2 * DC);
    cvt(sp_out_w, WSPOUT, (long)DC * DC4);
    cvt(proj_w, WPROJ, (long)DC * DC);

    // x (B,C,N) -> xf16 (B,N,C)
    k_transpose_f32_f16<<<dim3(DN / 32, DC / 32, DB), dim3(32, 8), 0, stream>>>(
        x, XF16, DC, DN, DN, DC, (long)DC * DN, (long)DN * DC);

    // qkv = xf @ qkv_w^T  -> (B,N,3C) f16
    k_gemm_wmma<<<dim3(DC3 / 64, DBN / 128, 1), 256, 0, stream>>>(
        XF16, WQKV, QKV, nullptr, DBN, DC3, DC, DC, DC, DC3,
        1, 0, 0, 0, 0, 0, 0, 0, DN);

    // per-channel 1/norm for q,k
    k_qk_norms<<<DB, 768, 0, stream>>>(QKV, RN);

    // q|k slice transposed to (B, 2C, N) for the Gram GEMM
    k_transpose_f16_f16<<<dim3(768 / 32, DN / 32, DB), dim3(32, 8), 0, stream>>>(
        QKV, QT, DN, 768, DC3, DN, (long)DN * DC3, (long)768 * DN);

    // Gram: per (b,h) 48x48, K = 4096
    k_gemm_wmma<<<dim3(1, 1, DB * DHEADS), 256, 0, stream>>>(
        QT, QT + (size_t)DC * DN, GRAM, nullptr, DHD, DHD, DN, DN, DN, DHD,
        DHEADS, (long)768 * DN, (long)DHD * DN, (long)768 * DN, (long)DHD * DN,
        (long)DHEADS * DHD * DHD, (long)DHD * DHD, 1, DN);

    // scaled softmax -> attn f16
    k_softmax48<<<DB * DHEADS, 64, 0, stream>>>(GRAM, RN, temperature, ATTN);

    // attened = v @ attn^T, per (b,h): (4096x48)x(48x48)
    k_gemm_wmma<<<dim3(1, DN / 128, DB * DHEADS), 256, 0, stream>>>(
        QKV + 2 * DC, ATTN, ATT, nullptr, DN, DHD, DHD, DC3, DHD, DC,
        DHEADS, (long)DN * DC3, (long)DHD, (long)DHEADS * DHD * DHD, (long)DHD * DHD,
        (long)DN * DC, (long)DHD, 0, DN);

    // dw1: 1x1 conv on v_ (read straight from qkv v-slice)
    k_gemm_wmma<<<dim3(DC / 64, DBN / 128, 1), 256, 0, stream>>>(
        QKV + 2 * DC, WDW1, CONV1, dw1_b, DBN, DC, DC, DC3, DC, DC,
        1, 0, 0, 0, 0, 0, 0, 0, DN);

    long totC = (long)DBN * DC;
    // dw2: depthwise 3x3
    k_dwconv<<<(unsigned)((totC + 255) / 256), 256, 0, stream>>>(
        CONV1, dw2_w, dw2_b, CONVX, DC, 3, 1, 1, totC);

    // cp_in: 1x1 conv C->C6 on attened
    k_gemm_wmma<<<dim3(1, DBN / 128, 1), 256, 0, stream>>>(
        ATT, WCPIN, T16, cp_in_b, DBN, DC6, DC, DC, DC, DC6,
        1, 0, 0, 0, 0, 0, 0, 0, DN);

    k_mean_n<<<DB, DC6, 0, stream>>>(T16, TMEAN, DC6);
    k_ci1<<<1, DB * DC6, 0, stream>>>(TMEAN, ci1_w, ci1_b, CI1v);

    long totC6 = (long)DBN * DC6;
    k_dwconv<<<(unsigned)((totC6 + 255) / 256), 256, 0, stream>>>(
        T16, ci2a_w, ci2a_b, CI2A, DC6, 3, 1, 1, totC6);
    k_dwconv<<<(unsigned)((totC6 + 255) / 256), 256, 0, stream>>>(
        CI2A, ci2b_w, ci2b_b, CI2B, DC6, 7, 9, 3, totC6);
    k_gemm_wmma<<<dim3(1, DBN / 128, 1), 256, 0, stream>>>(
        CI2B, WCI2C, CI2C, ci2c_b, DBN, DC6, DC6, DC6, DC6, DC6,
        1, 0, 0, 0, 0, 0, 0, 0, DN);

    k_gprod<<<(unsigned)((totC6 + 255) / 256), 256, 0, stream>>>(T16, CI1v, CI2C, GPROD, totC6);

    // cp_out: 1x1 conv C6->C
    k_gemm_wmma<<<dim3(DC / 64, DBN / 128, 1), 256, 0, stream>>>(
        GPROD, WCPOUT, CMAP, cp_out_b, DBN, DC, DC6, DC6, DC6, DC,
        1, 0, 0, 0, 0, 0, 0, 0, DN);

    k_add_inplace<<<(unsigned)((totC + 255) / 256), 256, 0, stream>>>(ATT, CMAP, totC);
    k_mean_n<<<DB, DC, 0, stream>>>(CMAP, CM, DC);

    // sp_in: 1x1 conv C->C2 on conv_x
    k_gemm_wmma<<<dim3(DC2 / 64, DBN / 128, 1), 256, 0, stream>>>(
        CONVX, WSPIN, SP, sp_in_b, DBN, DC2, DC, DC, DC, DC2,
        1, 0, 0, 0, 0, 0, 0, 0, DN);

    long totC2 = (long)DBN * DC2;
    k_dwconv<<<(unsigned)((totC2 + 255) / 256), 256, 0, stream>>>(
        SP, sp_dw_w, sp_dw_b, SPD, DC2, 3, 1, 1, totC2);

    long totC4 = (long)DBN * DC4;
    k_gated<<<(unsigned)((totC4 + 255) / 256), 256, 0, stream>>>(SPD, GT, totC4);

    // sp_out: 1x1 conv C4->C
    k_gemm_wmma<<<dim3(DC / 64, DBN / 128, 1), 256, 0, stream>>>(
        GT, WSPOUT, SMAP, sp_out_b, DBN, DC, DC4, DC4, DC4, DC,
        1, 0, 0, 0, 0, 0, 0, 0, DN);

    k_combine<<<(unsigned)((totC + 255) / 256), 256, 0, stream>>>(ATT, SMAP, CONVX, CM, totC);

    // final projection, f32 output transposed to (B,C,H,W)
    k_gemm_wmma<<<dim3(DC / 64, DBN / 128, 1), 256, 0, stream>>>(
        ATT, WPROJ, d_out, proj_b, DBN, DC, DC, DC, DC, 0,
        1, 0, 0, 0, 0, 0, 0, 2, DN);
}
